// TextEncoderMask_57234734186512
// MI455X (gfx1250) — matrix operand order repaired and checked by reading, not compiled, not run on hardware
//
#include <hip/hip_runtime.h>

// ---- problem constants (match reference) ----
#define BATCH 128
#define SEQ   64
#define DIM   512     // D
#define HID   1024    // H
#define EMBD  1024    // E
#define K3    3072    // E + 2H  (z-step K)
#define H3    3072    // 3*H

typedef __attribute__((ext_vector_type(16))) __bf16 v16bf;
typedef __attribute__((ext_vector_type(8)))  float  v8f;
typedef __attribute__((ext_vector_type(4)))  unsigned int v4u;

union Frag128 { v16bf v; v4u q[2]; };

__device__ __forceinline__ unsigned short f2bf(float f) {
  unsigned int u = __float_as_uint(f);
  unsigned int r = u + 0x7FFFu + ((u >> 16) & 1u);   // round-to-nearest-even
  return (unsigned short)(r >> 16);
}

// A-matrix (16x32 bf16, M x K) fragment per ISA 7.12.2. The 16 bf16 per lane are
// two contiguous 16B-aligned 16B chunks at byte offsets (2*k0 + 16*half) and +32.
__device__ __forceinline__ v16bf load_a_bf(const unsigned short* rowbase, int k0, int half) {
  const char* p = (const char*)rowbase + 2 * k0 + 16 * half;
  Frag128 f;
  f.q[0] = *(const v4u*)(p);
  f.q[1] = *(const v4u*)(p + 32);
  return f.v;
}

// B-matrix (32x16 bf16, K x N), B[k][n] = W[n][k]; lane n = q, same k order as A:
// two 16B chunks at byte offsets (2*k0 + 16*half) and +32 -> 2x (ds_|global_)load_b128.
__device__ __forceinline__ v16bf load_b_bf(const unsigned short* colbase, int k0, int half) {
  const char* p = (const char*)colbase + 2 * k0 + 16 * half;
  Frag128 f;
  f.q[0] = *(const v4u*)(p);
  f.q[1] = *(const v4u*)(p + 32);
  return f.v;
}

__device__ __forceinline__ v8f wmma_bf16(v16bf a, v16bf b, v8f c) {
  return __builtin_amdgcn_wmma_f32_16x16x32_bf16(false, a, false, b, (short)0, c, false, false);
}

// ---- CDNA5 async global->LDS copy (ASYNCcnt path), 16 bytes per lane ----
__device__ __forceinline__ void async_copy16(unsigned int lds_addr, const void* gptr) {
  asm volatile("global_load_async_to_lds_b128 %0, %1, off"
               :: "v"(lds_addr), "v"((unsigned long long)(uintptr_t)gptr)
               : "memory");
}
__device__ __forceinline__ void wait_async_and_barrier() {
  asm volatile("s_wait_asynccnt 0x0" ::: "memory");
  __syncthreads();
}

// ---------------- elementwise prep kernels ----------------
__global__ void cvt_bf16_kernel(const float* __restrict__ src,
                                unsigned short* __restrict__ dst, int n) {
  int i = blockIdx.x * blockDim.x + threadIdx.x;
  if (i < n) dst[i] = f2bf(src[i]);
}

// xe_bf[t][b][d] = bf16(emb[x[b][t]][d])   (time-major gather)
__global__ void gather_kernel(const int* __restrict__ x, const float* __restrict__ emb,
                              unsigned short* __restrict__ xe) {
  int i = blockIdx.x * blockDim.x + threadIdx.x;
  if (i >= SEQ * BATCH * DIM) return;
  int d = i % DIM;
  int r = i / DIM;
  int b = r % BATCH;
  int t = r / BATCH;
  xe[i] = f2bf(emb[(size_t)x[b * SEQ + t] * DIM + d]);
}

// ---------------- GRU step ----------------
// Block = 8 waves, one o-tile (16 h-cols); wave w handles batch tile b0 = 16*w.
// Weight strip (3 gates x 16 rows, w_ih + w_hh) staged once into LDS via async copy.
__global__ __launch_bounds__(256)
void gru_step_kernel(int s,
                     const unsigned short* __restrict__ xe_bf,   // [T,B,D] bf16
                     const unsigned short* __restrict__ wihF,
                     const unsigned short* __restrict__ whhF,
                     const float* __restrict__ bihF, const float* __restrict__ bhhF,
                     const unsigned short* __restrict__ wihB,
                     const unsigned short* __restrict__ whhB,
                     const float* __restrict__ bihB, const float* __restrict__ bhhB,
                     const int* __restrict__ lengths,
                     float* __restrict__ hstate_f,               // [2 dir][2 par][B,H] f32
                     unsigned short* __restrict__ hstate_bf,     // [2 dir][2 par][B,H] bf16
                     float* __restrict__ hf,  float* __restrict__ hb,   // [T,B,H] f32
                     unsigned short* __restrict__ hfbf,                 // [T,B,H] bf16
                     unsigned short* __restrict__ hbbf)
{
  extern __shared__ unsigned short smem[];          // 147456 B dynamic
  const int tid  = threadIdx.x;
  const int lane = tid & 31;
  const int q = lane & 15, half = lane >> 4;
  const int o0 = blockIdx.x * 16;       // h-column tile
  const int b0 = (tid >> 5) * 16;       // batch tile = wave id
  const int dir = blockIdx.z;
  const int t = dir ? (SEQ - 1 - s) : s;

  const unsigned short* wih = dir ? wihB : wihF;
  const unsigned short* whh = dir ? whhB : whhF;
  const float* bih = dir ? bihB : bihF;
  const float* bhh = dir ? bhhB : bhhF;

  // ---- stage weight strips into LDS (async, 16B chunks) ----
  // x-region: 48 rows x DIM bf16 = 49152 B = 3072 chunks
  for (int c = tid; c < 3072; c += 256) {
    int ob  = c * 16;                       // byte offset in region
    int row = ob >> 10;                     // DIM*2 = 1024 B per row
    int col = ob & 1023;
    int g = row >> 4, o = row & 15;
    const char* src = (const char*)(wih + (size_t)(g * HID + o0 + o) * DIM) + col;
    async_copy16((unsigned int)(uintptr_t)((char*)smem + ob), src);
  }
  // h-region: 48 rows x HID bf16 = 98304 B = 6144 chunks, base byte 49152
  for (int c = tid; c < 6144; c += 256) {
    int ob  = c * 16;
    int row = ob >> 11;                     // HID*2 = 2048 B per row
    int col = ob & 2047;
    int g = row >> 4, o = row & 15;
    const char* src = (const char*)(whh + (size_t)(g * HID + o0 + o) * HID) + col;
    async_copy16((unsigned int)(uintptr_t)((char*)smem + 49152 + ob), src);
  }
  wait_async_and_barrier();

  const size_t BH = (size_t)BATCH * HID;
  const float*          hprev_f  = hstate_f  + (size_t)(dir * 2 + (s & 1)) * BH;
  float*                hnext_f  = hstate_f  + (size_t)(dir * 2 + ((s + 1) & 1)) * BH;
  const unsigned short* hprev_bf = hstate_bf + (size_t)(dir * 2 + (s & 1)) * BH;
  unsigned short*       hnext_bf = hstate_bf + (size_t)(dir * 2 + ((s + 1) & 1)) * BH;
  float*          hout   = (dir ? hb : hf) + (size_t)t * BH;
  unsigned short* houtbf = (dir ? hbbf : hfbf) + (size_t)t * BH;

  v8f accR = {}; v8f accZ = {}; v8f accNX = {}; v8f accNH = {};

  // x-path: K over D; LDS rows (g*16+q)*DIM
  const unsigned short* aRow = xe_bf + ((size_t)t * BATCH + (b0 + q)) * DIM;
  const unsigned short* lRr = smem + (0 * 16 + q) * DIM;
  const unsigned short* lRz = smem + (1 * 16 + q) * DIM;
  const unsigned short* lRn = smem + (2 * 16 + q) * DIM;
  for (int k0 = 0; k0 < DIM; k0 += 32) {
    v16bf a  = load_a_bf(aRow, k0, half);
    v16bf br = load_b_bf(lRr, k0, half);
    v16bf bz = load_b_bf(lRz, k0, half);
    v16bf bn = load_b_bf(lRn, k0, half);
    accR  = wmma_bf16(a, br, accR);
    accZ  = wmma_bf16(a, bz, accZ);
    accNX = wmma_bf16(a, bn, accNX);
  }

  // h-path: K over H; LDS rows at element base 24576
  const unsigned short* hRow = hprev_bf + (size_t)(b0 + q) * HID;
  const unsigned short* lCr = smem + 24576 + (0 * 16 + q) * HID;
  const unsigned short* lCz = smem + 24576 + (1 * 16 + q) * HID;
  const unsigned short* lCn = smem + 24576 + (2 * 16 + q) * HID;
  for (int k0 = 0; k0 < HID; k0 += 32) {
    v16bf a  = load_a_bf(hRow, k0, half);
    v16bf br = load_b_bf(lCr, k0, half);
    v16bf bz = load_b_bf(lCz, k0, half);
    v16bf bn = load_b_bf(lCn, k0, half);
    accR  = wmma_bf16(a, br, accR);
    accZ  = wmma_bf16(a, bz, accZ);
    accNH = wmma_bf16(a, bn, accNH);
  }

  // gates + state update. C/D layout: lane n = q, VGPR v -> m = v + 8*half
  const int o = o0 + q;
  const float br_ = bih[o] + bhh[o];
  const float bz_ = bih[HID + o] + bhh[HID + o];
  const float bnx = bih[2 * HID + o];
  const float bnh = bhh[2 * HID + o];
#pragma unroll
  for (int v = 0; v < 8; ++v) {
    const int b = b0 + v + 8 * half;
    const bool mt = t < lengths[b];
    const float r  = 1.f / (1.f + __expf(-(accR[v] + br_)));
    const float zg = 1.f / (1.f + __expf(-(accZ[v] + bz_)));
    const float nn = tanhf(accNX[v] + bnx + r * (accNH[v] + bnh));
    const float hp = hprev_f[(size_t)b * HID + o];
    const float hnew = (1.f - zg) * nn + zg * hp;
    const float hnx = mt ? hnew : hp;
    const float out = mt ? hnew : 0.f;                 // padded outputs are zero
    hout[(size_t)b * HID + o]      = out;
    houtbf[(size_t)b * HID + o]    = f2bf(out);
    hnext_f[(size_t)b * HID + o]   = hnx;
    hnext_bf[(size_t)b * HID + o]  = f2bf(hnx);
  }
}

// ---------------- hidden_o ----------------
__global__ void hidden_kernel(const float* __restrict__ hf, const float* __restrict__ hb,
                              const int* __restrict__ lengths, float* __restrict__ out) {
  int idx = blockIdx.x * blockDim.x + threadIdx.x;
  if (idx >= BATCH * HID) return;
  int b = idx / HID, h = idx % HID;
  float acc = 0.f;
  for (int t = 0; t < SEQ; ++t) {
    size_t base = ((size_t)t * BATCH + b) * HID + h;
    acc += hf[base] + hb[base];
  }
  out[idx] = acc / (2.f * (float)lengths[b]);
}

// ---------------- z-step ----------------
// Block = 8 waves, one e-tile; wave w = batch tile. fc_w strip (mean rows e0..e0+15
// then logvar rows E+e0..E+e0+15, 32 rows x 3072 bf16 = 192KB) staged into LDS.
// A operands are pure bf16 (hf_bf / hb_bf mirrors + bf16 z-state ping-pong).
__global__ __launch_bounds__(256)
void z_step_kernel(int t,
                   const unsigned short* __restrict__ hfbf,    // [T,B,H] bf16
                   const unsigned short* __restrict__ hbbf,
                   const unsigned short* __restrict__ fcw_bf,  // [2E, 3072] bf16
                   const float* __restrict__ fcb,              // [2E]
                   const float* __restrict__ eps,              // [T,B,E]
                   unsigned short* __restrict__ zstate_bf,     // [2][B,E] bf16 ping-pong
                   float* __restrict__ means, float* __restrict__ logs,
                   float* __restrict__ zout)                   // each [B,T,E]
{
  extern __shared__ unsigned short smem[];          // 196608 B dynamic
  const int tid  = threadIdx.x;
  const int lane = tid & 31;
  const int q = lane & 15, half = lane >> 4;
  const int e0 = blockIdx.x * 16;
  const int b0 = (tid >> 5) * 16;

  // stage 32 rows x 6144 B = 12288 chunks
  for (int c = tid; c < 12288; c += 256) {
    int ob  = c * 16;
    int row = ob / 6144;                    // K3*2 B per row
    int col = ob - row * 6144;
    int grow = (row < 16) ? (e0 + row) : (EMBD + e0 + (row - 16));
    const char* src = (const char*)(fcw_bf + (size_t)grow * K3) + col;
    async_copy16((unsigned int)(uintptr_t)((char*)smem + ob), src);
  }
  wait_async_and_barrier();

  const unsigned short* zprev = zstate_bf + (size_t)(t & 1) * BATCH * EMBD;
  unsigned short*       znext = zstate_bf + (size_t)((t + 1) & 1) * BATCH * EMBD;

  const int brow = b0 + q;
  const unsigned short* hfRow = hfbf + ((size_t)t * BATCH + brow) * HID;
  const unsigned short* hbRow = hbbf + ((size_t)t * BATCH + brow) * HID;
  const unsigned short* zRow  = zprev + (size_t)brow * EMBD;
  const unsigned short* wM = smem;                           // mean rows, row = q
  const unsigned short* wL = smem + (size_t)16 * K3;         // logvar rows

  v8f accM = {}; v8f accL = {};
  for (int k0 = 0; k0 < K3; k0 += 32) {
    const unsigned short* src; int koff;
    if (k0 < HID)            { src = hfRow; koff = k0; }
    else if (k0 < 2 * HID)   { src = hbRow; koff = k0 - HID; }
    else                     { src = zRow;  koff = k0 - 2 * HID; }
    v16bf a  = load_a_bf(src, koff, half);
    v16bf bm = load_b_bf(wM + (size_t)q * K3, k0, half);
    v16bf bl = load_b_bf(wL + (size_t)q * K3, k0, half);
    accM = wmma_bf16(a, bm, accM);
    accL = wmma_bf16(a, bl, accL);
  }

  const int e = e0 + q;
  const float fbm = fcb[e], fbl = fcb[EMBD + e];
#pragma unroll
  for (int v = 0; v < 8; ++v) {
    const int b = b0 + v + 8 * half;
    const float mean = accM[v] + fbm;
    const float lg   = accL[v] + fbl;
    const float zt   = mean + eps[((size_t)t * BATCH + b) * EMBD + e] * __expf(0.5f * lg);
    const size_t oidx = ((size_t)b * SEQ + t) * EMBD + e;
    means[oidx] = mean;
    logs[oidx]  = lg;
    zout[oidx]  = zt;
    znext[(size_t)b * EMBD + e] = f2bf(zt);
  }
}

// ---------------- host-side orchestration ----------------
extern "C" void kernel_launch(void* const* d_in, const int* in_sizes, int n_in,
                              void* d_out, int out_size, void* d_ws, size_t ws_size,
                              hipStream_t stream) {
  const int*   x       = (const int*)d_in[0];
  const int*   lengths = (const int*)d_in[1];
  const float* emb     = (const float*)d_in[2];
  const float* w_ih_f  = (const float*)d_in[3];
  const float* w_hh_f  = (const float*)d_in[4];
  const float* b_ih_f  = (const float*)d_in[5];
  const float* b_hh_f  = (const float*)d_in[6];
  const float* w_ih_b  = (const float*)d_in[7];
  const float* w_hh_b  = (const float*)d_in[8];
  const float* b_ih_b  = (const float*)d_in[9];
  const float* b_hh_b  = (const float*)d_in[10];
  const float* fc_w    = (const float*)d_in[11];
  const float* fc_b    = (const float*)d_in[12];
  const float* eps     = (const float*)d_in[13];

  // workspace carve-up (all offsets 256B aligned by construction)
  char* ws = (char*)d_ws;
  size_t off = 0;
  unsigned short* wihf_bf = (unsigned short*)(ws + off); off += (size_t)H3 * DIM * 2;
  unsigned short* whhf_bf = (unsigned short*)(ws + off); off += (size_t)H3 * HID * 2;
  unsigned short* wihb_bf = (unsigned short*)(ws + off); off += (size_t)H3 * DIM * 2;
  unsigned short* whhb_bf = (unsigned short*)(ws + off); off += (size_t)H3 * HID * 2;
  unsigned short* fcw_bf  = (unsigned short*)(ws + off); off += (size_t)2 * EMBD * K3 * 2;
  unsigned short* xe_bf   = (unsigned short*)(ws + off); off += (size_t)SEQ * BATCH * DIM * 2;
  float*          hf      = (float*)(ws + off); off += (size_t)SEQ * BATCH * HID * 4;
  float*          hb      = (float*)(ws + off); off += (size_t)SEQ * BATCH * HID * 4;
  unsigned short* hfbf    = (unsigned short*)(ws + off); off += (size_t)SEQ * BATCH * HID * 2;
  unsigned short* hbbf    = (unsigned short*)(ws + off); off += (size_t)SEQ * BATCH * HID * 2;
  float*          hstate_f  = (float*)(ws + off); off += (size_t)4 * BATCH * HID * 4;
  unsigned short* hstate_bf = (unsigned short*)(ws + off); off += (size_t)4 * BATCH * HID * 2;
  unsigned short* zstate_bf = (unsigned short*)(ws + off); off += (size_t)2 * BATCH * EMBD * 2;

  // zero recurrent state (capture-safe)
  hipMemsetAsync(hstate_f, 0, (size_t)4 * BATCH * HID * 4, stream);
  hipMemsetAsync(hstate_bf, 0, (size_t)4 * BATCH * HID * 2, stream);
  hipMemsetAsync(zstate_bf, 0, (size_t)2 * BATCH * EMBD * 2, stream);

  // weight conversion to bf16
  auto cvt = [&](const float* s, unsigned short* d, int n) {
    cvt_bf16_kernel<<<(n + 255) / 256, 256, 0, stream>>>(s, d, n);
  };
  cvt(w_ih_f, wihf_bf, H3 * DIM);
  cvt(w_hh_f, whhf_bf, H3 * HID);
  cvt(w_ih_b, wihb_bf, H3 * DIM);
  cvt(w_hh_b, whhb_bf, H3 * HID);
  cvt(fc_w,   fcw_bf,  2 * EMBD * K3);

  // embedding gather (time-major, bf16)
  {
    int n = SEQ * BATCH * DIM;
    gather_kernel<<<(n + 255) / 256, 256, 0, stream>>>(x, emb, xe_bf);
  }

  // bidirectional GRU: 64 sequential steps, fwd+bwd fused per launch
  for (int s = 0; s < SEQ; ++s) {
    gru_step_kernel<<<dim3(HID / 16, 1, 2), 256, 147456, stream>>>(
        s, xe_bf, wihf_bf, whhf_bf, b_ih_f, b_hh_f,
        wihb_bf, whhb_bf, b_ih_b, b_hh_b, lengths,
        hstate_f, hstate_bf, hf, hb, hfbf, hbbf);
  }

  float* means_out  = (float*)d_out;
  float* logs_out   = means_out + (size_t)BATCH * SEQ * EMBD;
  float* z_out      = logs_out  + (size_t)BATCH * SEQ * EMBD;
  float* hidden_out = z_out     + (size_t)BATCH * SEQ * EMBD;

  {
    int n = BATCH * HID;
    hidden_kernel<<<(n + 255) / 256, 256, 0, stream>>>(hf, hb, lengths, hidden_out);
  }

  // z-scan: 64 sequential steps
  for (int t = 0; t < SEQ; ++t) {
    z_step_kernel<<<dim3(EMBD / 16, 1), 256, 196608, stream>>>(
        t, hfbf, hbbf, fcw_bf, fc_b, eps, zstate_bf, means_out, logs_out, z_out);
  }
}
